// EncoderBlock_32401233281245
// MI455X (gfx1250) — matrix operand order repaired and checked by reading, not compiled, not run on hardware
//
#include <hip/hip_runtime.h>
#include <hip/hip_bf16.h>

// ---------------------------------------------------------------------------
// Shapes (compile-time, from reference): B=2, S=2048, D=1024, H=16, DK=64, DFF=4096
// ---------------------------------------------------------------------------
constexpr int Bc  = 2;
constexpr int Sc  = 2048;
constexpr int Dc  = 1024;
constexpr int Hc  = 16;
constexpr int DKc = 64;
constexpr int DFFc = 4096;
constexpr int Mc  = Bc * Sc;      // 4096 rows

typedef __attribute__((ext_vector_type(16))) __bf16 v16bf;
typedef __attribute__((ext_vector_type(8)))  __bf16 v8bf;
typedef __attribute__((ext_vector_type(8)))  float  v8f;

union BF16x16 { v16bf v; v8bf h[2]; };

__device__ __forceinline__ v16bf join8(v8bf lo, v8bf hi) {
  BF16x16 u; u.h[0] = lo; u.h[1] = hi; return u.v;
}
__device__ __forceinline__ v8f zero8() {
  v8f z;
  #pragma unroll
  for (int i = 0; i < 8; ++i) z[i] = 0.0f;
  return z;
}

// gfx1250 async global->LDS copy (ASYNCcnt-tracked, bypasses VGPRs).
// vdst = per-lane LDS byte address, vaddr = per-lane 64-bit global address.
__device__ __forceinline__ void async_load_b128(unsigned lds_byte_addr, const void* gptr) {
  asm volatile("global_load_async_to_lds_b128 %0, %1, off"
               :: "v"(lds_byte_addr), "v"(gptr)
               : "memory");
}
__device__ __forceinline__ void wait_async0() {
  asm volatile("s_wait_asynccnt 0x0" ::: "memory");
}
// Generic shared-memory pointer -> LDS byte address (aperture rule: addr[31:0]).
__device__ __forceinline__ unsigned lds_addr(const void* p) {
  return (unsigned)(uintptr_t)p;
}

// ---------------------------------------------------------------------------
// f32 -> bf16 conversion (weights)
// ---------------------------------------------------------------------------
__global__ void f32_to_bf16(const float* __restrict__ src, __bf16* __restrict__ dst, int n) {
  for (int i = blockIdx.x * blockDim.x + threadIdx.x; i < n; i += gridDim.x * blockDim.x)
    dst[i] = (__bf16)src[i];
}

// ---------------------------------------------------------------------------
// LayerNorm: one block per row of D=1024; unbiased std (ddof=1), /(std+eps)
// ---------------------------------------------------------------------------
__global__ __launch_bounds__(256) void ln_kernel(const float* __restrict__ x,
                                                 const float* __restrict__ alpha,
                                                 const float* __restrict__ bias,
                                                 __bf16* __restrict__ out) {
  const int row = blockIdx.x;
  const int tid = threadIdx.x;
  const float* xr = x + (size_t)row * Dc;
  float v0[4], s = 0.f, s2 = 0.f;
  #pragma unroll
  for (int i = 0; i < 4; ++i) {
    float t = xr[tid + i * 256];
    v0[i] = t; s += t; s2 += t * t;
  }
  #pragma unroll
  for (int d = 1; d < 32; d <<= 1) { s += __shfl_xor(s, d); s2 += __shfl_xor(s2, d); }
  __shared__ float ws[8], ws2[8], stats[2];
  const int w = tid >> 5, lane = tid & 31;
  if (lane == 0) { ws[w] = s; ws2[w] = s2; }
  __syncthreads();
  if (tid == 0) {
    float ts = 0.f, ts2 = 0.f;
    #pragma unroll
    for (int i = 0; i < 8; ++i) { ts += ws[i]; ts2 += ws2[i]; }
    float mean = ts / (float)Dc;
    float var  = (ts2 - (float)Dc * mean * mean) / (float)(Dc - 1);
    float stdv = sqrtf(fmaxf(var, 0.f));
    stats[0] = mean;
    stats[1] = 1.0f / (stdv + 1e-6f);
  }
  __syncthreads();
  const float mean = stats[0], rstd = stats[1];
  __bf16* orow = out + (size_t)row * Dc;
  #pragma unroll
  for (int i = 0; i < 4; ++i) {
    int c = tid + i * 256;
    orow[c] = (__bf16)(alpha[c] * (v0[i] - mean) * rstd + bias[c]);
  }
}

// ---------------------------------------------------------------------------
// Generic bf16 WMMA GEMM: C[M,N] = A[M,K] * W[N,K]^T  (both K-contiguous)
// Block tile 128x128x32, 8 waves (4 in M x 2 in N), wave tile 32x64.
// Double-buffered LDS, filled by async global->LDS copies (ASYNCcnt).
// ---------------------------------------------------------------------------
enum { EP_SCATTER_QK = 0, EP_SCATTER_VT = 1, EP_RESID = 2, EP_BIAS_RELU = 3, EP_RESID_BIAS = 4 };

template <int N, int K, int EP>
__global__ __launch_bounds__(256) void gemm_bf16(const __bf16* __restrict__ A,
                                                 const __bf16* __restrict__ W,
                                                 __bf16* __restrict__ obf,
                                                 float* __restrict__ of32,
                                                 const float* __restrict__ resid,
                                                 const float* __restrict__ bias) {
  constexpr int LDT = 48;  // bf16 elements; 96B row stride, 16B aligned
  __shared__ __align__(16) __bf16 As[2][128 * LDT];
  __shared__ __align__(16) __bf16 Bs[2][128 * LDT];
  const int tid  = threadIdx.x;
  const int lane = tid & 31, w = tid >> 5;
  const int wm = w & 3, wn = w >> 2;
  const int bm = blockIdx.x * 128, bn = blockIdx.y * 128;
  const int l16 = lane & 15, lh = lane >> 4;

  // Tile-fill mapping: 128 rows x 4 chunks of 8 bf16 = 512 chunks; thread does
  // chunk tid (rows 0..63) and tid+256 (rows 64..127), same column chunk.
  const int r0 = tid >> 2, c8 = (tid & 3) * 8;

  auto stage = [&](int kt, int buf) {
    async_load_b128(lds_addr(&As[buf][r0 * LDT + c8]),
                    &A[(size_t)(bm + r0) * K + kt + c8]);
    async_load_b128(lds_addr(&As[buf][(r0 + 64) * LDT + c8]),
                    &A[(size_t)(bm + r0 + 64) * K + kt + c8]);
    async_load_b128(lds_addr(&Bs[buf][r0 * LDT + c8]),
                    &W[(size_t)(bn + r0) * K + kt + c8]);
    async_load_b128(lds_addr(&Bs[buf][(r0 + 64) * LDT + c8]),
                    &W[(size_t)(bn + r0 + 64) * K + kt + c8]);
  };

  v8f acc[2][4];
  #pragma unroll
  for (int i = 0; i < 2; ++i)
    #pragma unroll
    for (int j = 0; j < 4; ++j) acc[i][j] = zero8();

  constexpr int NK = K / 32;
  stage(0, 0);
  wait_async0();
  __syncthreads();

  for (int kt = 0; kt < NK; ++kt) {
    const int cur = kt & 1;
    if (kt + 1 < NK) stage((kt + 1) * 32, cur ^ 1);  // overlap with compute below

    v16bf af[2], bfr[4];
    #pragma unroll
    for (int mi = 0; mi < 2; ++mi) {        // A layout: lane halves hold K {0-7,16-23}/{8-15,24-31}
      int r = wm * 32 + mi * 16 + l16;
      af[mi] = join8(*(const v8bf*)&As[cur][r * LDT + 8 * lh],
                     *(const v8bf*)&As[cur][r * LDT + 16 + 8 * lh]);
    }
    #pragma unroll
    for (int ni = 0; ni < 4; ++ni) {        // B layout: lane n holds column n, K {0-7,16-23}/{8-15,24-31}
      int r = wn * 64 + ni * 16 + l16;
      bfr[ni] = join8(*(const v8bf*)&Bs[cur][r * LDT + 8 * lh],
                      *(const v8bf*)&Bs[cur][r * LDT + 16 + 8 * lh]);
    }
    #pragma unroll
    for (int mi = 0; mi < 2; ++mi)
      #pragma unroll
      for (int ni = 0; ni < 4; ++ni)
        acc[mi][ni] = __builtin_amdgcn_wmma_f32_16x16x32_bf16(
            false, af[mi], false, bfr[ni], (short)0, acc[mi][ni], false, false);

    wait_async0();       // next tile's async copies complete
    __syncthreads();     // all waves done reading `cur`, next buffer valid
  }

  // Epilogue. C frag layout: VGPR j = row (j + 8*(lane>=16)), col = lane%16.
  #pragma unroll
  for (int mi = 0; mi < 2; ++mi) {
    #pragma unroll
    for (int ni = 0; ni < 4; ++ni) {
      #pragma unroll
      for (int j = 0; j < 8; ++j) {
        const int r = bm + wm * 32 + mi * 16 + j + 8 * lh;   // global row (b*S+s)
        const int n = bn + wn * 64 + ni * 16 + l16;          // global col
        const float c = acc[mi][ni][j];
        if constexpr (EP == EP_SCATTER_QK) {
          int b = r >> 11, s = r & (Sc - 1);
          int h = n >> 6, d = n & (DKc - 1);
          obf[(((size_t)b * Hc + h) * Sc + s) * DKc + d] = (__bf16)c;
        } else if constexpr (EP == EP_SCATTER_VT) {
          int b = r >> 11, s = r & (Sc - 1);
          int h = n >> 6, d = n & (DKc - 1);
          obf[(((size_t)b * Hc + h) * DKc + d) * Sc + s] = (__bf16)c;
        } else if constexpr (EP == EP_RESID) {
          of32[(size_t)r * N + n] = resid[(size_t)r * N + n] + c;
        } else if constexpr (EP == EP_BIAS_RELU) {
          obf[(size_t)r * N + n] = (__bf16)fmaxf(c + bias[n], 0.f);
        } else {  // EP_RESID_BIAS
          of32[(size_t)r * N + n] = resid[(size_t)r * N + n] + c + bias[n];
        }
      }
    }
  }
}

// ---------------------------------------------------------------------------
// Flash-style attention: grid (S/128, B*H); 8 waves/block; wave owns 16 q-rows.
// q,k: [B,H,S,DK] bf16 ; vt: [B,H,DK,S] bf16 ; av out: [B,S,D] bf16
// ---------------------------------------------------------------------------
__global__ __launch_bounds__(256) void attn_kernel(const __bf16* __restrict__ q,
                                                   const __bf16* __restrict__ k,
                                                   const __bf16* __restrict__ vt,
                                                   const int* __restrict__ mask,
                                                   __bf16* __restrict__ av) {
  __shared__ __align__(16) __bf16 pl_all[8 * 16 * 32];  // per-wave private P tile
  const int tid = threadIdx.x, lane = tid & 31, w = tid >> 5;
  const int l16 = lane & 15, lh = lane >> 4;
  const int bh = blockIdx.y;
  const int b = bh >> 4, h = bh & (Hc - 1);
  const int qbase = blockIdx.x * 128 + w * 16;
  const __bf16* qh  = q  + (size_t)bh * Sc * DKc;
  const __bf16* kh  = k  + (size_t)bh * Sc * DKc;
  const __bf16* vth = vt + (size_t)bh * DKc * Sc;
  const int* mb = mask + b * Sc;
  __bf16* pl = pl_all + w * (16 * 32);

  // Q fragments (A-layout), kept resident
  v16bf aq[2];
  #pragma unroll
  for (int t = 0; t < 2; ++t) {
    const __bf16* qr = qh + (size_t)(qbase + l16) * DKc + t * 32;
    aq[t] = join8(*(const v8bf*)&qr[8 * lh], *(const v8bf*)&qr[16 + 8 * lh]);
  }

  v8f o[4];
  #pragma unroll
  for (int f = 0; f < 4; ++f) o[f] = zero8();
  float mrun[8], lrun[8];
  #pragma unroll
  for (int j = 0; j < 8; ++j) { mrun[j] = -3.0e38f; lrun[j] = 0.f; }

  for (int kk = 0; kk < Sc; kk += 32) {
    // scores for 32 keys: two 16x16 fragments, DK=64 -> 2 K-steps each
    v8f s0 = zero8(), s1 = zero8();
    #pragma unroll
    for (int t = 0; t < 2; ++t) {
      const __bf16* kr0 = kh + (size_t)(kk + l16) * DKc + t * 32 + 8 * lh;
      v16bf bk0 = join8(*(const v8bf*)&kr0[0], *(const v8bf*)&kr0[16]);
      s0 = __builtin_amdgcn_wmma_f32_16x16x32_bf16(false, aq[t], false, bk0, (short)0, s0, false, false);
      const __bf16* kr1 = kh + (size_t)(kk + 16 + l16) * DKc + t * 32 + 8 * lh;
      v16bf bk1 = join8(*(const v8bf*)&kr1[0], *(const v8bf*)&kr1[16]);
      s1 = __builtin_amdgcn_wmma_f32_16x16x32_bf16(false, aq[t], false, bk1, (short)0, s1, false, false);
    }
    const float sc = 0.125f;  // 1/sqrt(64)
    const int m0 = mb[kk + l16], m1 = mb[kk + 16 + l16];
    float f0[8], f1[8], a8[8];
    #pragma unroll
    for (int j = 0; j < 8; ++j) {
      f0[j] = m0 ? s0[j] * sc : -1e9f;
      f1[j] = m1 ? s1[j] * sc : -1e9f;
    }
    #pragma unroll
    for (int j = 0; j < 8; ++j) {
      float mj = fmaxf(f0[j], f1[j]);
      #pragma unroll
      for (int d2 = 1; d2 < 16; d2 <<= 1) mj = fmaxf(mj, __shfl_xor(mj, d2));
      float mnew  = fmaxf(mrun[j], mj);
      float alpha = __expf(mrun[j] - mnew);
      float p0 = __expf(f0[j] - mnew);
      float p1 = __expf(f1[j] - mnew);
      float rs = p0 + p1;
      #pragma unroll
      for (int d2 = 1; d2 < 16; d2 <<= 1) rs += __shfl_xor(rs, d2);
      lrun[j] = lrun[j] * alpha + rs;
      mrun[j] = mnew;
      a8[j] = alpha;
      const int prow = j + 8 * lh;                   // C-layout row
      pl[prow * 32 + l16]      = (__bf16)p0;
      pl[prow * 32 + 16 + l16] = (__bf16)p1;
    }
    #pragma unroll
    for (int f = 0; f < 4; ++f)
      #pragma unroll
      for (int j = 0; j < 8; ++j) o[f][j] *= a8[j];
    // C-layout -> A-layout transpose via per-wave LDS; DS ops are in-order per wave
    asm volatile("s_wait_dscnt 0x0" ::: "memory");
    const __bf16* pr = pl + l16 * 32;
    v16bf pa = join8(*(const v8bf*)&pr[8 * lh], *(const v8bf*)&pr[16 + 8 * lh]);
    #pragma unroll
    for (int f = 0; f < 4; ++f) {
      const __bf16* vr = vth + (size_t)(f * 16 + l16) * Sc + kk + 8 * lh;
      v16bf bv = join8(*(const v8bf*)&vr[0], *(const v8bf*)&vr[16]);
      o[f] = __builtin_amdgcn_wmma_f32_16x16x32_bf16(false, pa, false, bv, (short)0, o[f], false, false);
    }
  }

  float inv[8];
  #pragma unroll
  for (int j = 0; j < 8; ++j) inv[j] = 1.0f / lrun[j];
  #pragma unroll
  for (int f = 0; f < 4; ++f) {
    #pragma unroll
    for (int j = 0; j < 8; ++j) {
      const int srow = qbase + j + 8 * lh;
      const int d = h * DKc + f * 16 + l16;
      av[(size_t)(b * Sc + srow) * Dc + d] = (__bf16)(o[f][j] * inv[j]);
    }
  }
}

// ---------------------------------------------------------------------------
// Launch
// ---------------------------------------------------------------------------
extern "C" void kernel_launch(void* const* d_in, const int* in_sizes, int n_in,
                              void* d_out, int out_size, void* d_ws, size_t ws_size,
                              hipStream_t stream) {
  (void)in_sizes; (void)n_in; (void)out_size; (void)ws_size;
  const float* x      = (const float*)d_in[0];
  const int*   mask   = (const int*)  d_in[1];
  const float* wq     = (const float*)d_in[2];
  const float* wk     = (const float*)d_in[3];
  const float* wv     = (const float*)d_in[4];
  const float* wo     = (const float*)d_in[5];
  const float* w1     = (const float*)d_in[6];
  const float* b1     = (const float*)d_in[7];
  const float* w2     = (const float*)d_in[8];
  const float* b2     = (const float*)d_in[9];
  const float* alpha1 = (const float*)d_in[10];
  const float* bias1  = (const float*)d_in[11];
  const float* alpha2 = (const float*)d_in[12];
  const float* bias2  = (const float*)d_in[13];
  float* out = (float*)d_out;

  char* ws = (char*)d_ws;
  size_t off = 0;
  auto carve = [&](size_t bytes) -> char* {
    char* p = ws + off;
    off += (bytes + 255) & ~(size_t)255;
    return p;
  };
  __bf16* wq_b = (__bf16*)carve((size_t)Dc * Dc * 2);
  __bf16* wk_b = (__bf16*)carve((size_t)Dc * Dc * 2);
  __bf16* wv_b = (__bf16*)carve((size_t)Dc * Dc * 2);
  __bf16* wo_b = (__bf16*)carve((size_t)Dc * Dc * 2);
  __bf16* w1_b = (__bf16*)carve((size_t)DFFc * Dc * 2);
  __bf16* w2_b = (__bf16*)carve((size_t)Dc * DFFc * 2);
  __bf16* xn   = (__bf16*)carve((size_t)Mc * Dc * 2);     // reused for xn2
  __bf16* qb   = (__bf16*)carve((size_t)Mc * Dc * 2);
  __bf16* kb   = (__bf16*)carve((size_t)Mc * Dc * 2);
  __bf16* vtb  = (__bf16*)carve((size_t)Mc * Dc * 2);
  __bf16* avb  = (__bf16*)carve((size_t)Mc * Dc * 2);
  float*  x1   = (float*) carve((size_t)Mc * Dc * 4);
  __bf16* hb   = (__bf16*)carve((size_t)Mc * DFFc * 2);

  // weight precision conversion
  f32_to_bf16<<<2048, 256, 0, stream>>>(wq, wq_b, Dc * Dc);
  f32_to_bf16<<<2048, 256, 0, stream>>>(wk, wk_b, Dc * Dc);
  f32_to_bf16<<<2048, 256, 0, stream>>>(wv, wv_b, Dc * Dc);
  f32_to_bf16<<<2048, 256, 0, stream>>>(wo, wo_b, Dc * Dc);
  f32_to_bf16<<<2048, 256, 0, stream>>>(w1, w1_b, DFFc * Dc);
  f32_to_bf16<<<2048, 256, 0, stream>>>(w2, w2_b, Dc * DFFc);

  // --- residual 1: pre-norm + attention ---
  ln_kernel<<<Mc, 256, 0, stream>>>(x, alpha1, bias1, xn);
  dim3 gD(Mc / 128, Dc / 128);
  gemm_bf16<Dc, Dc, EP_SCATTER_QK><<<gD, 256, 0, stream>>>(xn, wq_b, qb, nullptr, nullptr, nullptr);
  gemm_bf16<Dc, Dc, EP_SCATTER_QK><<<gD, 256, 0, stream>>>(xn, wk_b, kb, nullptr, nullptr, nullptr);
  gemm_bf16<Dc, Dc, EP_SCATTER_VT><<<gD, 256, 0, stream>>>(xn, wv_b, vtb, nullptr, nullptr, nullptr);
  attn_kernel<<<dim3(Sc / 128, Bc * Hc), 256, 0, stream>>>(qb, kb, vtb, mask, avb);
  gemm_bf16<Dc, Dc, EP_RESID><<<gD, 256, 0, stream>>>(avb, wo_b, nullptr, x1, x, nullptr);

  // --- residual 2: pre-norm + FFN ---
  ln_kernel<<<Mc, 256, 0, stream>>>(x1, alpha2, bias2, xn);
  gemm_bf16<DFFc, Dc, EP_BIAS_RELU><<<dim3(Mc / 128, DFFc / 128), 256, 0, stream>>>(
      xn, w1_b, hb, nullptr, nullptr, b1);
  gemm_bf16<Dc, DFFc, EP_RESID_BIAS><<<gD, 256, 0, stream>>>(hb, w2_b, nullptr, out, x1, b2);
}
